// FullyConnected_35244501631569
// MI455X (gfx1250) — matrix operand order repaired and checked by reading, not compiled, not run on hardware
//
#include <hip/hip_runtime.h>

#define IN_SIZE 4096
#define OUT_SIZE 4096
#define BATCH 256

typedef __attribute__((ext_vector_type(16))) __bf16 v16bf;
typedef __attribute__((ext_vector_type(8)))  float  v8f;

// ---------------------------------------------------------------------------
// Phase 0: zero the dense-W workspace (must happen every call; ws is not
// re-poisoned between replays and we accumulate atomically into it).
// ---------------------------------------------------------------------------
__global__ void zero_w_kernel(float4* __restrict__ w, int n4) {
    int i = blockIdx.x * blockDim.x + threadIdx.x;
    if (i < n4) w[i] = float4{0.f, 0.f, 0.f, 0.f};
}

// ---------------------------------------------------------------------------
// Phase 1: COO scatter-add into dense W[IN_SIZE][OUT_SIZE] (row-major).
// Duplicate (row,col) pairs must accumulate -> float atomic add. W lives in
// the 192MB L2, so global_atomic_add_f32 resolves on-chip.
// ---------------------------------------------------------------------------
__global__ void scatter_coo_kernel(const long long* __restrict__ idx,
                                   const float* __restrict__ val,
                                   float* __restrict__ W, int nnz) {
    int i = blockIdx.x * blockDim.x + threadIdx.x;
    if (i >= nnz) return;
    long long r = idx[2 * i];       // input index  (0..IN_SIZE)
    long long c = idx[2 * i + 1];   // output index (0..OUT_SIZE)
    unsafeAtomicAdd(&W[(size_t)r * OUT_SIZE + (size_t)c], val[i]);
}

// ---------------------------------------------------------------------------
// Phase 2: Z = relu(X @ W + bias) via bf16x3 split-precision WMMA.
//   x = hi + lo with hi = bf16(x), lo = bf16(x - hi)
//   A*B ~= Ahi*Bhi + Ahi*Blo + Alo*Bhi   (f32 accumulate)
// One wave: 1 N-tile (16 cols) x 4 M-tiles (64 rows). 12 wmma per K-step.
// ---------------------------------------------------------------------------
__launch_bounds__(256, 1)
__global__ void gemm_bf16x3_kernel(const float* __restrict__ X,
                                   const float* __restrict__ W,
                                   const float* __restrict__ bias,
                                   float* __restrict__ out) {
    const int lane = threadIdx.x & 31;
    const int wave = threadIdx.x >> 5;
    const int nt    = blockIdx.x * 8 + wave;   // N tile index (0..255)
    const int nbase = nt * 16;
    const int mbase = blockIdx.y * 64;         // 4 M tiles of 16 rows

    const int nl = lane & 15;                  // column within tile / row id
    const int hl = lane >> 4;                  // lane-half select

    v8f acc[4] = {{}, {}, {}, {}};

    for (int kb = 0; kb < IN_SIZE; kb += 32) {
        // ---- B tile: W[kb..kb+31][nbase..nbase+15], bf16 hi/lo split ----
        // K offset: hl*8 then +16 block (matches A) ; N = nl
        const float* wp = W + (size_t)(kb + hl * 8) * OUT_SIZE + nbase + nl;
        v16bf bhi, blo;
#pragma unroll
        for (int e = 0; e < 16; ++e) {
            float f   = wp[(size_t)(e + (e & 8)) * OUT_SIZE];
            __bf16 h  = (__bf16)f;
            bhi[e]    = h;
            blo[e]    = (__bf16)(f - (float)h);
        }

#pragma unroll
        for (int t = 0; t < 4; ++t) {
            // ---- A tile t: X[mbase+t*16 .. +15][kb..kb+31] ----
            // lane row = nl (both halves); K offset: hl*8 then +16 block
            const float* xp = X + (size_t)(mbase + t * 16 + nl) * IN_SIZE
                                + kb + hl * 8;
            v16bf ahi, alo;
#pragma unroll
            for (int e = 0; e < 8; ++e) {
                float f  = xp[e];
                __bf16 h = (__bf16)f;
                ahi[e]   = h;
                alo[e]   = (__bf16)(f - (float)h);
            }
#pragma unroll
            for (int e = 0; e < 8; ++e) {
                float f      = xp[16 + e];
                __bf16 h     = (__bf16)f;
                ahi[8 + e]   = h;
                alo[8 + e]   = (__bf16)(f - (float)h);
            }

            acc[t] = __builtin_amdgcn_wmma_f32_16x16x32_bf16(
                false, ahi, false, bhi, (short)0, acc[t], false, false);
            acc[t] = __builtin_amdgcn_wmma_f32_16x16x32_bf16(
                false, ahi, false, blo, (short)0, acc[t], false, false);
            acc[t] = __builtin_amdgcn_wmma_f32_16x16x32_bf16(
                false, alo, false, bhi, (short)0, acc[t], false, false);
        }
    }

    // ---- epilogue: bias + relu, store. C/D: row = hl*8 + r, col = nl ----
    const float b = bias[nbase + nl];
#pragma unroll
    for (int t = 0; t < 4; ++t) {
#pragma unroll
        for (int r = 0; r < 8; ++r) {
            int row = mbase + t * 16 + hl * 8 + r;
            float z = acc[t][r] + b;
            out[(size_t)row * OUT_SIZE + nbase + nl] = fmaxf(z, 0.f);
        }
    }
}

// ---------------------------------------------------------------------------
extern "C" void kernel_launch(void* const* d_in, const int* in_sizes, int n_in,
                              void* d_out, int out_size, void* d_ws, size_t ws_size,
                              hipStream_t stream) {
    const float*     X    = (const float*)d_in[0];
    const long long* idx  = (const long long*)d_in[1];
    const float*     val  = (const float*)d_in[2];
    const float*     bias = (const float*)d_in[3];
    float*           W    = (float*)d_ws;      // 4096*4096*4 = 64 MB
    float*           out  = (float*)d_out;

    const int nnz = in_sizes[2];

    // Phase 0: zero dense W
    const int n4 = (IN_SIZE * OUT_SIZE) / 4;
    zero_w_kernel<<<(n4 + 255) / 256, 256, 0, stream>>>((float4*)W, n4);

    // Phase 1: scatter-add COO values
    scatter_coo_kernel<<<(nnz + 255) / 256, 256, 0, stream>>>(idx, val, W, nnz);

    // Phase 2: WMMA GEMM + bias + relu
    dim3 grid(OUT_SIZE / 16 / 8, BATCH / 64);  // (32, 4), 256 thr = 8 waves
    gemm_bf16x3_kernel<<<grid, dim3(256, 1, 1), 0, stream>>>(X, W, bias, out);
    (void)n_in; (void)out_size; (void)ws_size;
}
